// GAT_22634477650567
// MI455X (gfx1250) — matrix operand order repaired and checked by reading, not compiled, not run on hardware
//
#include <hip/hip_runtime.h>
#include <hip/hip_bf16.h>

// GAT, 2 layers, N=4096 D=256 H=8 O=32, MI455X (gfx1250, wave32, WMMA).
//
//  - proj_kernel:  Wh = X @ W per head via v_wmma_f32_16x16x32_bf16,
//                  fused src/dst = Wh . a epilogue. Wh stored transposed
//                  ([H][O][N] bf16) so attention B-fragments are contiguous.
//  - attn_kernel:  flash-style: per (head, 16-row block) stream 16x32 score
//                  tiles: e = lrelu(src_i + dst_j), masked exp, probabilities
//                  -> bf16 A-frag, accumulate P @ Wh with 2 WMMAs per chunk.
//                  adj mask tiles are prefetched with double-buffered
//                  GLOBAL_LOAD_ASYNC_TO_LDS_B128 (ASYNCcnt) when available.
//                  Epilogue: softmax-normalize, ELU, residual, bf16 copy.

#define GN 4096
#define GD 256
#define GH 8
#define GO 32

#define GLOBAL_AS __attribute__((address_space(1)))
#define LDS_AS    __attribute__((address_space(3)))

#if defined(__has_builtin)
#if __has_builtin(__builtin_amdgcn_global_load_async_to_lds_b128)
#define HAVE_ASYNC_LDS 1
#endif
#endif
#ifndef HAVE_ASYNC_LDS
#define HAVE_ASYNC_LDS 0
#endif

typedef __attribute__((ext_vector_type(16))) __bf16 v16bf;
typedef __attribute__((ext_vector_type(8)))  float  v8f;
typedef __attribute__((ext_vector_type(4)))  int    v4i;

union U16B { uint4 u[2]; __bf16 h[16]; };
union U8B  { uint4 u;    __bf16 h[8];  };

__device__ __forceinline__ v16bf ld16bf(const __bf16* p) {
  U16B t;
  t.u[0] = *(const uint4*)(p);
  t.u[1] = *(const uint4*)(p + 16);
  v16bf v;
#pragma unroll
  for (int i = 0; i < 16; ++i) v[i] = t.h[i];
  return v;
}

__device__ __forceinline__ v8f wmma_bf16(v16bf a, v16bf b, v8f c) {
  // (neg_a, A, neg_b, B, c_mod, C, reuse_a, reuse_b)
  return __builtin_amdgcn_wmma_f32_16x16x32_bf16(false, a, false, b, (short)0, c,
                                                 false, false);
}

// ---------------------------------------------------------------- converts
__global__ void cvt_f32_bf16(const float* __restrict__ in,
                             __bf16* __restrict__ out, int n) {
  int i = blockIdx.x * blockDim.x + threadIdx.x;
  if (i < n) out[i] = (__bf16)in[i];
}

// W: [GH][GD][GO] f32  ->  Wt: [GH][GO][GD] bf16 (transposed for B-frags)
__global__ void cvt_w_transpose(const float* __restrict__ W,
                                __bf16* __restrict__ Wt) {
  int i = blockIdx.x * blockDim.x + threadIdx.x;
  if (i >= GH * GD * GO) return;
  int o = i % GO;
  int d = (i / GO) % GD;
  int h = i / (GO * GD);
  Wt[(h * GO + o) * GD + d] = (__bf16)W[i];
}

// ---------------------------------------------------------------- projection
// One wave per (head, 16-row block). Computes 16x32 tile of Wh = X @ W_h,
// writes Wh transposed bf16, and fused src/dst dot products.
__global__ void __launch_bounds__(256)
proj_kernel(const __bf16* __restrict__ xb,   // [GN][GD]
            const __bf16* __restrict__ Wt,   // [GH][GO][GD]
            const float*  __restrict__ av,   // [GH][2*GO]
            __bf16* __restrict__ Wht,        // [GH][GO][GN]
            float* __restrict__ srcv,        // [GH][GN]
            float* __restrict__ dstv)        // [GH][GN]
{
  const int lane = threadIdx.x & 31;
  const int wid  = blockIdx.x * (blockDim.x >> 5) + (threadIdx.x >> 5);
  const int h    = wid / (GN / 16);
  const int rb   = (wid % (GN / 16)) * 16;
  const int hi   = lane >> 4;   // half-wave
  const int l16  = lane & 15;

  v8f acc0 = {}; v8f acc1 = {};
  const __bf16* arow  = xb + (size_t)(rb + l16) * GD;
  const __bf16* b0row = Wt + (size_t)(h * GO + l16) * GD;
  const __bf16* b1row = Wt + (size_t)(h * GO + 16 + l16) * GD;

#pragma unroll 2
  for (int kb = 0; kb < GD; kb += 32) {
    // A-frag (16-bit A layout): lane holds row l16, K runs {0..7,16..23}(+8*hi)
    v16bf A;
    {
      const int k0 = kb + hi * 8;
      U16B t;
      t.u[0] = *(const uint4*)(arow + k0);
      t.u[1] = *(const uint4*)(arow + k0 + 16);
#pragma unroll
      for (int i = 0; i < 16; ++i) A[i] = t.h[i];
    }
    // B-frags: lane = column o, K runs {0..7,16..23}(+8*hi)
    const int kk = kb + hi * 8;
    v16bf B0 = ld16bf(b0row + kk);
    v16bf B1 = ld16bf(b1row + kk);
    acc0 = wmma_bf16(A, B0, acc0);
    acc1 = wmma_bf16(A, B1, acc1);
  }

  // Store Wh transposed bf16: C layout -> lane writes 8 contiguous n values.
  {
    U8B t0, t1;
#pragma unroll
    for (int r = 0; r < 8; ++r) {
      t0.h[r] = (__bf16)acc0[r];
      t1.h[r] = (__bf16)acc1[r];
    }
    *(uint4*)(Wht + (size_t)(h * GO + l16) * GN + rb + hi * 8)      = t0.u;
    *(uint4*)(Wht + (size_t)(h * GO + 16 + l16) * GN + rb + hi * 8) = t1.u;
  }

  // Fused src/dst: src[n] = sum_o Wh[n][o]*a[o]; dst with a[O..2O).
  const float* ah = av + h * 2 * GO;
  const float as0 = ah[l16],       as1 = ah[16 + l16];
  const float ad0 = ah[GO + l16],  ad1 = ah[GO + 16 + l16];
  float s[8], d[8];
#pragma unroll
  for (int r = 0; r < 8; ++r) {
    s[r] = acc0[r] * as0 + acc1[r] * as1;
    d[r] = acc0[r] * ad0 + acc1[r] * ad1;
  }
#pragma unroll
  for (int m = 1; m < 16; m <<= 1) {
#pragma unroll
    for (int r = 0; r < 8; ++r) {
      s[r] += __shfl_xor(s[r], m, 32);
      d[r] += __shfl_xor(d[r], m, 32);
    }
  }
  if (l16 == 0) {
#pragma unroll
    for (int r = 0; r < 8; ++r) {
      srcv[h * GN + rb + hi * 8 + r] = s[r];
      dstv[h * GN + rb + hi * 8 + r] = d[r];
    }
  }
}

// ---------------------------------------------------------------- attention
// One wave per (head, 16-row block). Streams K in chunks of 32.
__global__ void __launch_bounds__(256)
attn_kernel(const float*  __restrict__ srcv,  // [GH][GN]
            const float*  __restrict__ dstv,  // [GH][GN]
            const __bf16* __restrict__ Wht,   // [GH][GO][GN]
            const int*    __restrict__ adj,   // [GN][GN]
            const float*  __restrict__ resid, // [GN][GD]
            float*        __restrict__ out,   // [GN][GD]
            __bf16*       __restrict__ outb)  // [GN][GD] bf16 or null
{
  __shared__ float rs[8][16];
#if HAVE_ASYNC_LDS
  __shared__ int adjbuf[8][2][16 * 32];   // 32 KB: per-wave double buffer
#endif
  const int lane = threadIdx.x & 31;
  const int w    = threadIdx.x >> 5;
  const int wid  = blockIdx.x * (blockDim.x >> 5) + w;
  const int h    = wid / (GN / 16);
  const int rb   = (wid % (GN / 16)) * 16;
  const int hi   = lane >> 4;
  const int l16  = lane & 15;

  const float   si    = srcv[h * GN + rb + l16];
  const int*    arow  = adj  + (size_t)(rb + l16) * GN;
  const float*  drow  = dstv + (size_t)h * GN;
  const __bf16* b0row = Wht + (size_t)(h * GO + l16) * GN;
  const __bf16* b1row = Wht + (size_t)(h * GO + 16 + l16) * GN;

  v8f acc0 = {}; v8f acc1 = {};
  float rsum = 0.f;

#if HAVE_ASYNC_LDS
  // Async prefetch of one 16x32 adj tile: lane (l16,hi) covers row l16,
  // columns [hi*16, hi*16+16) -> 4 x b128 async ops per wave-chunk.
  const int lslot = l16 * 32 + hi * 16;   // int index within a buffer
  auto adj_prefetch = [&](int jb, int buf) {
    const int* g = arow + jb + hi * 16;
    int* l = &adjbuf[w][buf][lslot];
#pragma unroll
    for (int k = 0; k < 4; ++k) {
      __builtin_amdgcn_global_load_async_to_lds_b128(
          (GLOBAL_AS v4i*)(g + k * 4), (LDS_AS v4i*)(l + k * 4), 0, 0);
    }
  };
  adj_prefetch(0, 0);
#endif

  for (int jb = 0; jb < GN; jb += 32) {
    const int j0 = jb + hi * 8;   // A-layout: runs [j0,j0+8) and [j0+16,j0+24)
    union { int4   v[4]; int   s[16]; } mm;
    union { float4 v[4]; float s[16]; } dd;

#if HAVE_ASYNC_LDS
    const int buf = (jb >> 5) & 1;
    if (jb + 32 < GN) {
      adj_prefetch(jb + 32, buf ^ 1);
      asm volatile("s_wait_asynccnt 0x4" ::: "memory");  // current tile landed
    } else {
      asm volatile("s_wait_asynccnt 0x0" ::: "memory");
    }
    const int* lb = &adjbuf[w][buf][l16 * 32 + hi * 8];
    mm.v[0] = *(const int4*)(lb);
    mm.v[1] = *(const int4*)(lb + 4);
    mm.v[2] = *(const int4*)(lb + 16);
    mm.v[3] = *(const int4*)(lb + 20);
#else
    mm.v[0] = *(const int4*)(arow + j0);
    mm.v[1] = *(const int4*)(arow + j0 + 4);
    mm.v[2] = *(const int4*)(arow + j0 + 16);
    mm.v[3] = *(const int4*)(arow + j0 + 20);
#endif
    dd.v[0] = *(const float4*)(drow + j0);
    dd.v[1] = *(const float4*)(drow + j0 + 4);
    dd.v[2] = *(const float4*)(drow + j0 + 16);
    dd.v[3] = *(const float4*)(drow + j0 + 20);

    v16bf A;
#pragma unroll
    for (int e = 0; e < 16; ++e) {
      float ev = si + dd.s[e];
      ev = (ev >= 0.f) ? ev : 0.2f * ev;                  // leaky_relu(0.2)
      const float p = (mm.s[e] > 0) ? __expf(ev) : 0.f;   // mask == exp(-inf)=0
      rsum += p;
      A[e] = (__bf16)p;
    }

    v16bf B0 = ld16bf(b0row + jb + hi * 8);
    v16bf B1 = ld16bf(b1row + jb + hi * 8);
    acc0 = wmma_bf16(A, B0, acc0);
    acc1 = wmma_bf16(A, B1, acc1);
  }

  // rowsum: lanes l and l+16 cover disjoint columns of the same row
  rsum += __shfl_xor(rsum, 16, 32);
  if (hi == 0) rs[w][l16] = rsum;
  asm volatile("s_wait_dscnt 0" ::: "memory");  // wave-local LDS exchange

#pragma unroll
  for (int r = 0; r < 8; ++r) {
    const int   n   = rb + r + hi * 8;
    const float inv = 1.0f / rs[w][r + hi * 8];
    float v0 = acc0[r] * inv;
    float v1 = acc1[r] * inv;
    v0 = (v0 > 0.f) ? v0 : (__expf(v0) - 1.f);    // elu
    v1 = (v1 > 0.f) ? v1 : (__expf(v1) - 1.f);
    const int c0 = h * GO + l16;
    const float o0 = v0 + resid[(size_t)n * GD + c0];
    const float o1 = v1 + resid[(size_t)n * GD + c0 + 16];
    out[(size_t)n * GD + c0]      = o0;
    out[(size_t)n * GD + c0 + 16] = o1;
    if (outb) {
      outb[(size_t)n * GD + c0]      = (__bf16)o0;
      outb[(size_t)n * GD + c0 + 16] = (__bf16)o1;
    }
  }
}

// ---------------------------------------------------------------- launch
extern "C" void kernel_launch(void* const* d_in, const int* in_sizes, int n_in,
                              void* d_out, int out_size, void* d_ws, size_t ws_size,
                              hipStream_t stream) {
  (void)in_sizes; (void)n_in; (void)out_size; (void)ws_size;
  const float* x   = (const float*)d_in[0];
  const int*   adj = (const int*)d_in[1];
  const float* W1  = (const float*)d_in[2];
  const float* a1  = (const float*)d_in[3];
  const float* W2  = (const float*)d_in[4];
  const float* a2  = (const float*)d_in[5];
  float* out = (float*)d_out;

  char* ws = (char*)d_ws;
  size_t off = 0;
  auto alloc = [&](size_t bytes) -> void* {
    off = (off + 255) & ~(size_t)255;
    void* p = ws + off;
    off += bytes;
    return p;
  };
  __bf16* xb   = (__bf16*)alloc((size_t)GN * GD * 2);
  __bf16* W1t  = (__bf16*)alloc((size_t)GH * GO * GD * 2);
  __bf16* W2t  = (__bf16*)alloc((size_t)GH * GO * GD * 2);
  __bf16* Wht  = (__bf16*)alloc((size_t)GH * GO * GN * 2);
  float*  srcb = (float*)alloc((size_t)GH * GN * 4);
  float*  dstb = (float*)alloc((size_t)GH * GN * 4);
  float*  h1   = (float*)alloc((size_t)GN * GD * 4);
  __bf16* h1b  = (__bf16*)alloc((size_t)GN * GD * 2);

  cvt_f32_bf16<<<(GN * GD + 255) / 256, 256, 0, stream>>>(x, xb, GN * GD);
  cvt_w_transpose<<<(GH * GD * GO + 255) / 256, 256, 0, stream>>>(W1, W1t);
  cvt_w_transpose<<<(GH * GD * GO + 255) / 256, 256, 0, stream>>>(W2, W2t);

  const int waves = GH * (GN / 16);    // 2048 waves
  dim3 blk(256), grd(waves / 8);       // 8 waves/block -> 256 blocks

  // Layer 1
  proj_kernel<<<grd, blk, 0, stream>>>(xb, W1t, a1, Wht, srcb, dstb);
  attn_kernel<<<grd, blk, 0, stream>>>(srcb, dstb, Wht, adj, x, h1, h1b);
  // Layer 2 (W2 is [H][H*O=256][O], same shape as W1 transposed path)
  proj_kernel<<<grd, blk, 0, stream>>>(h1b, W2t, a2, Wht, srcb, dstb);
  attn_kernel<<<grd, blk, 0, stream>>>(srcb, dstb, Wht, adj, h1, out,
                                       (__bf16*)nullptr);
}
